// LightRowTransformer_33217277067678
// MI455X (gfx1250) — compile-verified, matrix operand order checked by reading, not run on hardware
//
#include <hip/hip_runtime.h>
#include <math.h>

typedef __attribute__((ext_vector_type(16))) _Float16 v16h;
typedef __attribute__((ext_vector_type(8)))  _Float16 v8h;
typedef __attribute__((ext_vector_type(8)))  float    v8f;

#define B_   128
#define C_   8
#define H_   144
#define W_   144
#define CW_  1152
#define WP_  148
#define OG_  2
#define TW_  5
#define NCLS_ 6
#define DT_  256
#define ITC_ 5760
#define EPS_ 1e-5f

// ---------------- workspace layout (bytes) ----------------
static constexpr size_t OFF_SUM = 0;                       // 1 float
static constexpr size_t OFF_AE  = 256;                     // 256 f32
static constexpr size_t OFF_CE  = OFF_AE + 1024;
static constexpr size_t OFF_AC  = OFF_CE + 1024;
static constexpr size_t OFF_CC  = OFF_AC + 1024;
static constexpr size_t OFF_B2E = OFF_CC + 1024;           // 2 f32
static constexpr size_t OFF_B2C = OFF_B2E + 256;           // 144 f32
static constexpr size_t OFF_W2E = 8192;                    // 16*256 f16
static constexpr size_t OFF_W2C = OFF_W2E + 16*256*2;      // 144*256 f16
static constexpr size_t OFF_W1E = OFF_W2C + 144*256*2;     // 256*1152 f16
static constexpr size_t OFF_W1C = OFF_W1E + 256*CW_*2;
static constexpr size_t OFF_XT  = OFF_W1C + 256*CW_*2;     // B*H*CW f16
static constexpr size_t OFF_HTE = OFF_XT  + (size_t)B_*H_*CW_*2;  // B*H*256 f16
static constexpr size_t OFF_HTC = OFF_HTE + (size_t)B_*H_*256*2;
static constexpr size_t OFF_XP  = OFF_HTC + (size_t)B_*H_*256*2;  // B*C*H*WP f32

// ---------------- tiny init ----------------
__global__ void k_init(float* s){ if (threadIdx.x==0 && blockIdx.x==0) *s = 0.f; }

// ---------------- f32 -> f16 weight convert ----------------
__global__ void k_cvt_w1(const float* __restrict__ w1e, const float* __restrict__ w1c,
                         _Float16* __restrict__ oe, _Float16* __restrict__ oc){
  int n = 256*CW_;
  for (int i = blockIdx.x*blockDim.x + threadIdx.x; i < n; i += gridDim.x*blockDim.x){
    oe[i] = (_Float16)w1e[i];
    oc[i] = (_Float16)w1c[i];
  }
}

// ---------------- fold BN affine into second matmul ----------------
__global__ void k_precomp(const float* b1e,const float* ge,const float* bbe,const float* me,const float* ve,
                          const float* w2e,const float* b2e,
                          const float* b1c,const float* gc,const float* bbc,const float* mc,const float* vc,
                          const float* w2c,const float* b2c,
                          float* aE, float* cE, float* aC, float* cC,
                          _Float16* w2effE, _Float16* w2effC, float* b2effE, float* b2effC){
  int t = threadIdx.x;                              // 256 threads
  float ae = ge[t]*rsqrtf(ve[t]+EPS_);
  aE[t] = ae; cE[t] = ae*b1e[t] + bbe[t] - ae*me[t];
  float ac = gc[t]*rsqrtf(vc[t]+EPS_);
  aC[t] = ac; cC[t] = ac*b1c[t] + bbc[t] - ac*mc[t];
  __syncthreads();
  for (int i = t; i < 16*256; i += 256){
    int r = i>>8, o = i&255;
    w2effE[i] = (_Float16)((r < 2) ? w2e[r*256+o]*aE[o] : 0.f);
  }
  for (int i = t; i < 144*256; i += 256){
    int r = i>>8, o = i&255;
    w2effC[i] = (_Float16)(w2c[r*256+o]*aC[o]);
  }
  if (t < 2){ float s=b2e[t]; for (int o=0;o<256;++o) s += w2e[t*256+o]*cE[o]; b2effE[t]=s; }
  if (t < 144){ float s=b2c[t]; for (int o=0;o<256;++o) s += w2c[t*256+o]*cC[o]; b2effC[t]=s; }
}

// ---------------- zero-pad x -> xp (width 148) ----------------
__global__ void k_pad(const float* __restrict__ x, float* __restrict__ xp){
  long n = (long)B_*C_*H_*WP_;
  for (long i = (long)blockIdx.x*blockDim.x + threadIdx.x; i < n; i += (long)gridDim.x*blockDim.x){
    int wp = (int)(i % WP_); long pre = i / WP_;
    xp[i] = (wp >= OG_ && wp < W_+OG_) ? x[pre*W_ + (wp-OG_)] : 0.f;
  }
}

// ---------------- pack (b,c,h,w) -> xt[b][h][c*W+w] f16 ----------------
__global__ void k_pack(const float* __restrict__ src, int srcW, int woff, _Float16* __restrict__ xt){
  long n = (long)B_*H_*CW_;
  for (long i = (long)blockIdx.x*blockDim.x + threadIdx.x; i < n; i += (long)gridDim.x*blockDim.x){
    int k = (int)(i % CW_); long bh = i / CW_;
    int h = (int)(bh % H_); long b = bh / H_;
    int c = k / W_, w = k % W_;
    xt[i] = (_Float16)src[(((b*C_+c)*H_+h)*(long)srcW) + w + woff];
  }
}

// ---------------- GEMM1: hT[b][h][o] = (W1 @ row)^T, f16 WMMA ----------------
// K outermost, double-buffered 144x32 B-slab in LDS, 9 accumulators per wave,
// one barrier per K-chunk, 72 wmmas per block per sync.
__global__ void __launch_bounds__(256) k_gemm1(const _Float16* __restrict__ xt,
                                               const _Float16* __restrict__ w1e,
                                               const _Float16* __restrict__ w1c,
                                               _Float16* __restrict__ hTe,
                                               _Float16* __restrict__ hTc){
  __shared__ alignas(16) _Float16 Bt[2][144][32];    // 2 x 9 KB
  int b = blockIdx.x;
  int Mbase = blockIdx.y * 128;
  const _Float16* W1 = blockIdx.z ? w1c : w1e;
  _Float16* hT = blockIdx.z ? hTc : hTe;
  int tid = threadIdx.x, wave = tid >> 5, lane = tid & 31;
  int lrow = lane & 15, hi = lane >> 4;
  int Mw = Mbase + wave*16;
  const _Float16* xtb = xt + (long)b*H_*CW_;

  v8f zero = {};
  v8f acc[9];
  #pragma unroll
  for (int i = 0; i < 9; ++i) acc[i] = zero;

  // fill coords: 2304 dwords = 144 rows x 16 dwords, 9 per thread
  int frow[9], fkp[9];
  #pragma unroll
  for (int j = 0; j < 9; ++j){
    int e = tid + j*256;
    frow[j] = e >> 4; fkp[j] = (e & 15) << 1;
  }

  uint32_t fr[9];
  #pragma unroll
  for (int j = 0; j < 9; ++j)
    fr[j] = *(const uint32_t*)&xtb[frow[j]*CW_ + fkp[j]];          // kc = 0
  #pragma unroll
  for (int j = 0; j < 9; ++j)
    *(uint32_t*)&Bt[0][frow[j]][fkp[j]] = fr[j];
  __syncthreads();

  for (int kc = 0; kc < 36; ++kc){
    int cur = kc & 1;
    if (kc + 1 < 36){
      #pragma unroll
      for (int j = 0; j < 9; ++j)
        fr[j] = *(const uint32_t*)&xtb[frow[j]*CW_ + (kc+1)*32 + fkp[j]];
      if (kc + 2 < 36)
        __builtin_prefetch(&xtb[frow[0]*CW_ + (kc+2)*32 + fkp[0]], 0, 1);
    }
    int k0 = kc*32 + hi*8;
    union { v16h v; v8h h[2]; } A;
    A.h[0] = *(const v8h*)&W1[(Mw + lrow)*CW_ + k0];
    A.h[1] = *(const v8h*)&W1[(Mw + lrow)*CW_ + k0 + 16];
    #pragma unroll
    for (int nt = 0; nt < 9; ++nt){
      union { v16h v; v8h h[2]; } Bf;
      Bf.h[0] = *(const v8h*)&Bt[cur][nt*16 + lrow][hi*8];
      Bf.h[1] = *(const v8h*)&Bt[cur][nt*16 + lrow][hi*8 + 16];
      acc[nt] = __builtin_amdgcn_wmma_f32_16x16x32_f16(false, A.v, false, Bf.v,
                                                       (short)0, acc[nt], false, false);
    }
    if (kc + 1 < 36){
      #pragma unroll
      for (int j = 0; j < 9; ++j)
        *(uint32_t*)&Bt[cur^1][frow[j]][fkp[j]] = fr[j];
    }
    __syncthreads();                                 // single barrier per K-chunk
  }

  #pragma unroll
  for (int nt = 0; nt < 9; ++nt){
    int n  = nt*16 + lrow;
    int mb = Mw + hi*8;
    v8h o;
    #pragma unroll
    for (int r = 0; r < 8; ++r) o[r] = (_Float16)acc[nt][r];
    *(v8h*)&hT[((long)b*H_ + n)*256 + mb] = o;       // lane's 8 consecutive M
  }
}

// ---------------- GEMM2: logits = w2eff @ h + b2eff, direct into d_out ----------------
__global__ void __launch_bounds__(256) k_gemm2(const _Float16* __restrict__ hTe,
                                               const _Float16* __restrict__ hTc,
                                               const _Float16* __restrict__ w2effE,
                                               const _Float16* __restrict__ w2effC,
                                               const float* __restrict__ b2effE,
                                               const float* __restrict__ b2effC,
                                               float* __restrict__ outE,
                                               float* __restrict__ outC){
  int b = blockIdx.x, head = blockIdx.y;
  int tid = threadIdx.x, wave = tid >> 5, lane = tid & 31;
  int lrow = lane & 15, hi = lane >> 4;
  const _Float16* hT = head ? hTc : hTe;
  if (head == 0){
    for (int nt = wave; nt < 9; nt += 8){
      v8f acc = {};
      for (int kc = 0; kc < 8; ++kc){
        int k0 = kc*32 + hi*8;
        union { v16h v; v8h h[2]; } A, Bf;
        A.h[0] = *(const v8h*)&w2effE[lrow*256 + k0];
        A.h[1] = *(const v8h*)&w2effE[lrow*256 + k0 + 16];
        int n = nt*16 + lrow;
        Bf.h[0] = *(const v8h*)&hT[((long)b*H_ + n)*256 + k0];
        Bf.h[1] = *(const v8h*)&hT[((long)b*H_ + n)*256 + k0 + 16];
        acc = __builtin_amdgcn_wmma_f32_16x16x32_f16(false, A.v, false, Bf.v,
                                                     (short)0, acc, false, false);
      }
      if (hi == 0){                                   // only m=0,1 valid
        int n = nt*16 + lrow;
        float2 o2 = make_float2(acc[0] + b2effE[0], acc[1] + b2effE[1]);
        *(float2*)&outE[((long)b*H_ + n)*2] = o2;
      }
    }
  } else {
    for (int t = wave; t < 81; t += 8){
      int mt = t/9, nt = t%9;
      v8f acc = {};
      for (int kc = 0; kc < 8; ++kc){
        int k0 = kc*32 + hi*8;
        union { v16h v; v8h h[2]; } A, Bf;
        int m = mt*16 + lrow;
        A.h[0] = *(const v8h*)&w2effC[m*256 + k0];
        A.h[1] = *(const v8h*)&w2effC[m*256 + k0 + 16];
        int n = nt*16 + lrow;
        Bf.h[0] = *(const v8h*)&hT[((long)b*H_ + n)*256 + k0];
        Bf.h[1] = *(const v8h*)&hT[((long)b*H_ + n)*256 + k0 + 16];
        acc = __builtin_amdgcn_wmma_f32_16x16x32_f16(false, A.v, false, Bf.v,
                                                     (short)0, acc, false, false);
      }
      int n = nt*16 + lrow;
      int mb = mt*16 + hi*8;
      float4 s0 = make_float4(acc[0]+b2effC[mb+0], acc[1]+b2effC[mb+1],
                              acc[2]+b2effC[mb+2], acc[3]+b2effC[mb+3]);
      float4 s1 = make_float4(acc[4]+b2effC[mb+4], acc[5]+b2effC[mb+5],
                              acc[6]+b2effC[mb+6], acc[7]+b2effC[mb+7]);
      float* dst = &outC[(((long)b*H_ + n)*W_) + mb];
      *(float4*)dst = s0; *(float4*)(dst+4) = s1;
    }
  }
}

// ---------------- softmax over 2 (+ optional mean accumulation) ----------------
__global__ void k_smax_ext(float* __restrict__ o, float* sum, int doSum){
  int i = blockIdx.x*blockDim.x + threadIdx.x;
  if (i >= B_*H_) return;
  float x0 = o[i*2], x1 = o[i*2+1];
  float m = fmaxf(x0, x1);
  float e0 = expf(x0-m), e1 = expf(x1-m);
  float inv = 1.f/(e0+e1);
  o[i*2] = e0*inv; o[i*2+1] = e1*inv;
  if (doSum) atomicAdd(sum, e0*inv);
}

// ---------------- softmax over 144, one wave per row ----------------
__global__ void k_smax_cls(float* __restrict__ o){
  int wave = threadIdx.x >> 5, lane = threadIdx.x & 31;
  int row = blockIdx.x*8 + wave;
  if (row >= B_*H_) return;
  float* p = o + (long)row*W_;
  float v[5]; float mx = -1e30f;
  #pragma unroll
  for (int j = 0; j < 5; ++j){ int i = lane + 32*j; v[j] = (i < W_) ? p[i] : -1e30f; mx = fmaxf(mx, v[j]); }
  for (int off = 16; off; off >>= 1) mx = fmaxf(mx, __shfl_xor(mx, off, 32));
  float s = 0.f;
  #pragma unroll
  for (int j = 0; j < 5; ++j){ int i = lane + 32*j; v[j] = (i < W_) ? expf(v[j]-mx) : 0.f; s += v[j]; }
  for (int off = 16; off; off >>= 1) s += __shfl_xor(s, off, 32);
  float inv = 1.f/s;
  #pragma unroll
  for (int j = 0; j < 5; ++j){ int i = lane + 32*j; if (i < W_) p[i] = v[j]*inv; }
}

// ---------------- LN over 6 tokens x 256 (block-uniform) ----------------
__device__ inline void ln6(const float* src, float* dst, const float* g, const float* bb,
                           float* mu6, float* rs6, int t){
  if (t < NCLS_){
    float s = 0.f; for (int d = 0; d < DT_; ++d) s += src[t*DT_+d];
    float mu = s/DT_;
    float vv = 0.f; for (int d = 0; d < DT_; ++d){ float e = src[t*DT_+d]-mu; vv += e*e; }
    mu6[t] = mu; rs6[t] = rsqrtf(vv/DT_ + EPS_);
  }
  __syncthreads();
  for (int i = t; i < NCLS_*DT_; i += 256){
    int tk = i/DT_, d = i%DT_;
    dst[i] = (src[i]-mu6[tk])*rs6[tk]*g[d] + bb[d];
  }
  __syncthreads();
}

// ---------------- predicated refine: gather -> transformer -> scatter ----------------
__global__ void __launch_bounds__(256) k_refine(const float* __restrict__ clsProb,
                                                const float* __restrict__ sumP0,
                                                float* __restrict__ xp,
    const float* tok_w, const float* tok_b, const float* emb,
    const float* ln1_g, const float* ln1_b, const float* qkv_w,
    const float* out_w, const float* out_b,
    const float* ln2_g, const float* ln2_b,
    const float* ff_w1, const float* ff_b1, const float* ff_w2, const float* ff_b2,
    const float* lnf_g, const float* lnf_b, const float* fin_w, const float* fin_b){
  if (*sumP0 <= 0.3f * (float)(B_*H_)) return;        // lax.cond predicate
  __shared__ float P[ITC_];
  __shared__ int   idx0[H_];
  __shared__ float tks[NCLS_*DT_];
  __shared__ float z[NCLS_*DT_];
  __shared__ float qkv[NCLS_*192];
  __shared__ float att[2*NCLS_*NCLS_];
  __shared__ float oat[NCLS_*64];
  __shared__ float ff[NCLS_*512];
  __shared__ float mu6[NCLS_], rs6[NCLS_];
  int b = blockIdx.x, t = threadIdx.x;
  if (t < H_){
    const float* row = clsProb + ((long)b*H_ + t)*W_;
    float best = row[0]; int bi = 0;
    for (int w = 1; w < W_; ++w){ float val = row[w]; if (val > best){ best = val; bi = w; } }
    idx0[t] = bi;
  }
  __syncthreads();
  for (int k = t; k < ITC_; k += 256){
    int c = k/(H_*TW_), rem = k%(H_*TW_), h = rem/TW_, tt = rem%TW_;
    P[k] = xp[(((long)b*C_+c)*H_+h)*WP_ + idx0[h] + tt];
  }
  __syncthreads();
  {
    float s = tok_b[t];
    for (int k = 0; k < ITC_; ++k) s += P[k]*tok_w[(long)k*DT_ + t];
    for (int i = 0; i < NCLS_; ++i) tks[i*DT_+t] = s + emb[i*DT_+t];
  }
  __syncthreads();
  ln6(tks, z, ln1_g, ln1_b, mu6, rs6, t);
  for (int i = t; i < NCLS_*192; i += 256){
    int tk = i/192, d = i%192;
    float s = 0.f;
    for (int kk = 0; kk < DT_; ++kk) s += z[tk*DT_+kk]*qkv_w[kk*192+d];
    qkv[i] = s;
  }
  __syncthreads();
  if (t < 2*NCLS_*NCLS_){
    int hh = t/36, rem = t%36, i = rem/NCLS_, j = rem%NCLS_;
    float s = 0.f;
    for (int dd = 0; dd < 32; ++dd) s += qkv[i*192 + hh*32+dd]*qkv[j*192 + 64 + hh*32+dd];
    att[t] = s * 0.17677669529663687f;                // 32^-0.5
  }
  __syncthreads();
  if (t < 2*NCLS_){
    int base = t*NCLS_;
    float m = att[base]; for (int j = 1; j < NCLS_; ++j) m = fmaxf(m, att[base+j]);
    float s = 0.f;
    for (int j = 0; j < NCLS_; ++j){ att[base+j] = expf(att[base+j]-m); s += att[base+j]; }
    float inv = 1.f/s;
    for (int j = 0; j < NCLS_; ++j) att[base+j] *= inv;
  }
  __syncthreads();
  if (t < NCLS_*64){
    int i = t/64, rem = t%64, hh = rem/32, dd = rem%32;
    float s = 0.f;
    for (int j = 0; j < NCLS_; ++j) s += att[(hh*NCLS_+i)*NCLS_+j]*qkv[j*192 + 128 + hh*32+dd];
    oat[t] = s;
  }
  __syncthreads();
  for (int i = t; i < NCLS_*DT_; i += 256){
    int tk = i/DT_, d = i%DT_;
    float s = out_b[d];
    for (int dd = 0; dd < 64; ++dd) s += oat[tk*64+dd]*out_w[dd*DT_+d];
    tks[i] += s;
  }
  __syncthreads();
  ln6(tks, z, ln2_g, ln2_b, mu6, rs6, t);
  for (int i = t; i < NCLS_*512; i += 256){
    int tk = i/512, d = i%512;
    float s = ff_b1[d];
    for (int kk = 0; kk < DT_; ++kk) s += z[tk*DT_+kk]*ff_w1[kk*512+d];
    ff[i] = 0.5f*s*(1.f + erff(s*0.7071067811865476f));   // exact gelu
  }
  __syncthreads();
  for (int i = t; i < NCLS_*DT_; i += 256){
    int tk = i/DT_, d = i%DT_;
    float s = ff_b2[d];
    for (int kk = 0; kk < 512; ++kk) s += ff[tk*512+kk]*ff_w2[kk*DT_+d];
    tks[i] += s;
  }
  __syncthreads();
  ln6(tks, z, lnf_g, lnf_b, mu6, rs6, t);
  // only class NCLS-1 survives the reference's repeated scatter
  const float* zf = &z[(NCLS_-1)*DT_];
  for (int o = t; o < ITC_; o += 256){
    float s = fin_b[o];
    for (int d = 0; d < DT_; ++d) s += zf[d]*fin_w[(long)d*ITC_ + o];
    int c = o/(H_*TW_), rem = o%(H_*TW_), h = rem/TW_, tt = rem%TW_;
    xp[(((long)b*C_+c)*H_+h)*WP_ + idx0[h] + tt] = s;
  }
}

extern "C" void kernel_launch(void* const* d_in, const int* in_sizes, int n_in,
                              void* d_out, int out_size, void* d_ws, size_t ws_size,
                              hipStream_t stream) {
  (void)in_sizes; (void)n_in; (void)out_size; (void)ws_size;
  const float* x      = (const float*)d_in[0];
  const float* ext_w1 = (const float*)d_in[1];
  const float* ext_b1 = (const float*)d_in[2];
  const float* ext_g  = (const float*)d_in[3];
  const float* ext_bb = (const float*)d_in[4];
  const float* ext_m  = (const float*)d_in[5];
  const float* ext_v  = (const float*)d_in[6];
  const float* ext_w2 = (const float*)d_in[7];
  const float* ext_b2 = (const float*)d_in[8];
  const float* cls_w1 = (const float*)d_in[9];
  const float* cls_b1 = (const float*)d_in[10];
  const float* cls_g  = (const float*)d_in[11];
  const float* cls_bb = (const float*)d_in[12];
  const float* cls_m  = (const float*)d_in[13];
  const float* cls_v  = (const float*)d_in[14];
  const float* cls_w2 = (const float*)d_in[15];
  const float* cls_b2 = (const float*)d_in[16];
  const float* tok_w  = (const float*)d_in[17];
  const float* tok_b  = (const float*)d_in[18];
  const float* emb    = (const float*)d_in[19];
  const float* ln1_g  = (const float*)d_in[20];
  const float* ln1_b  = (const float*)d_in[21];
  const float* qkv_w  = (const float*)d_in[22];
  const float* out_w  = (const float*)d_in[23];
  const float* out_b  = (const float*)d_in[24];
  const float* ln2_g  = (const float*)d_in[25];
  const float* ln2_b  = (const float*)d_in[26];
  const float* ff_w1  = (const float*)d_in[27];
  const float* ff_b1  = (const float*)d_in[28];
  const float* ff_w2  = (const float*)d_in[29];
  const float* ff_b2  = (const float*)d_in[30];
  const float* lnf_g  = (const float*)d_in[31];
  const float* lnf_b  = (const float*)d_in[32];
  const float* fin_w  = (const float*)d_in[33];
  const float* fin_b  = (const float*)d_in[34];

  char* ws = (char*)d_ws;
  float*    sum    = (float*)(ws + OFF_SUM);
  float*    aE     = (float*)(ws + OFF_AE);
  float*    cE     = (float*)(ws + OFF_CE);
  float*    aC     = (float*)(ws + OFF_AC);
  float*    cC     = (float*)(ws + OFF_CC);
  float*    b2effE = (float*)(ws + OFF_B2E);
  float*    b2effC = (float*)(ws + OFF_B2C);
  _Float16* w2effE = (_Float16*)(ws + OFF_W2E);
  _Float16* w2effC = (_Float16*)(ws + OFF_W2C);
  _Float16* w1Eh   = (_Float16*)(ws + OFF_W1E);
  _Float16* w1Ch   = (_Float16*)(ws + OFF_W1C);
  _Float16* xt     = (_Float16*)(ws + OFF_XT);
  _Float16* hTe    = (_Float16*)(ws + OFF_HTE);
  _Float16* hTc    = (_Float16*)(ws + OFF_HTC);
  float*    xp     = (float*)(ws + OFF_XP);

  float* out   = (float*)d_out;
  float* extO  = out;                              // (B,H,2)
  float* clsO  = out + (size_t)B_*H_*2;            // (B,H,144)
  float* ext2O = clsO + (size_t)B_*H_*W_;
  float* cls2O = ext2O + (size_t)B_*H_*2;

  k_init<<<1, 1, 0, stream>>>(sum);
  k_cvt_w1<<<512, 256, 0, stream>>>(ext_w1, cls_w1, w1Eh, w1Ch);
  k_precomp<<<1, 256, 0, stream>>>(ext_b1, ext_g, ext_bb, ext_m, ext_v, ext_w2, ext_b2,
                                   cls_b1, cls_g, cls_bb, cls_m, cls_v, cls_w2, cls_b2,
                                   aE, cE, aC, cC, w2effE, w2effC, b2effE, b2effC);
  k_pad<<<2048, 256, 0, stream>>>(x, xp);

  // ---- pass 1: heads on x ----
  k_pack<<<4096, 256, 0, stream>>>(x, W_, 0, xt);
  dim3 g1(B_, 2, 2);
  k_gemm1<<<g1, 256, 0, stream>>>(xt, w1Eh, w1Ch, hTe, hTc);
  dim3 g2(B_, 2);
  k_gemm2<<<g2, 256, 0, stream>>>(hTe, hTc, w2effE, w2effC, b2effE, b2effC, extO, clsO);
  k_smax_ext<<<(B_*H_ + 255)/256, 256, 0, stream>>>(extO, sum, 1);
  k_smax_cls<<<(B_*H_ + 7)/8, 256, 0, stream>>>(clsO);

  // ---- predicated refine (mutates xp iff mean(ext p0) > 0.3) ----
  k_refine<<<B_, 256, 0, stream>>>(clsO, sum, xp,
                                   tok_w, tok_b, emb, ln1_g, ln1_b, qkv_w, out_w, out_b,
                                   ln2_g, ln2_b, ff_w1, ff_b1, ff_w2, ff_b2,
                                   lnf_g, lnf_b, fin_w, fin_b);

  // ---- pass 2: heads on x2 = xp[..., OG:W+OG] ----
  k_pack<<<4096, 256, 0, stream>>>(xp, WP_, OG_, xt);
  k_gemm1<<<g1, 256, 0, stream>>>(xt, w1Eh, w1Ch, hTe, hTc);
  k_gemm2<<<g2, 256, 0, stream>>>(hTe, hTc, w2effE, w2effC, b2effE, b2effC, ext2O, cls2O);
  k_smax_ext<<<(B_*H_ + 255)/256, 256, 0, stream>>>(ext2O, sum, 0);
  k_smax_cls<<<(B_*H_ + 7)/8, 256, 0, stream>>>(cls2O);
}
